// GATEncoder_46746424050053
// MI455X (gfx1250) — matrix operand order repaired and checked by reading, not compiled, not run on hardware
//
#include <hip/hip_runtime.h>
#include <hip/hip_bf16.h>

typedef __attribute__((ext_vector_type(16))) __bf16 v16bf;
typedef __attribute__((ext_vector_type(8)))  float  v8f;

#define NODE_DIM 128
#define HC       256   // HEADS*OUT_DIM
#define HEADS    4
#define ODIM     64
#define EDIM     16
#define KSTEPS   4     // 128 / 32

// ---------- helpers: order-preserving float<->uint for atomicMax on floats ----------
__device__ __forceinline__ unsigned encf(float f) {
  unsigned u = __float_as_uint(f);
  return (u & 0x80000000u) ? ~u : (u | 0x80000000u);
}
__device__ __forceinline__ float decf(unsigned u) {
  u = (u & 0x80000000u) ? (u & 0x7FFFFFFFu) : ~u;
  return __uint_as_float(u);
}

// ---------- K0: per-dst sums of edge_attr + counts (for self-loop fill 'mean') ----------
__global__ void k_loop_accum(const float* __restrict__ edge_attr,
                             const int* __restrict__ dst,
                             float* __restrict__ loop_sum,
                             float* __restrict__ cnt, int E) {
  int idx = blockIdx.x * blockDim.x + threadIdx.x;
  if (idx >= E * EDIM) return;
  int e = idx >> 4, d = idx & 15;
  int n = dst[e];
  atomicAdd(&loop_sum[n * EDIM + d], edge_attr[idx]);
  if (d == 0) atomicAdd(&cnt[n], 1.0f);
}

// ---------- K1: V[d][h] = sum_c W_e[d][h*64+c] * att_edge[h][c]  (folds edge GEMM) ----------
__global__ void k_precompute_v(const float* __restrict__ W_e,
                               const float* __restrict__ att_edge,
                               float* __restrict__ v) {
  int t = threadIdx.x;            // 64 threads
  int d = t >> 2, hd = t & 3;
  float s = 0.f;
  for (int c = 0; c < ODIM; ++c)
    s += W_e[d * HC + hd * ODIM + c] * att_edge[hd * ODIM + c];
  v[d * HEADS + hd] = s;
}

// ---------- K2a: pre-swizzle W into B-operand bf16 hi/lo register layout ----------
// Element index: (((tn*KSTEPS + ks)*32 + lane)*16 + e); B layout (ISA 7.12.2):
// e<8 -> K = half*8 + e; e>=8 -> K = 16 + half*8 + (e-8), N = lane&15.
__global__ void k_swizzle_w(const float* __restrict__ W,
                            __bf16* __restrict__ wsw_hi, __bf16* __restrict__ wsw_lo) {
  int idx = blockIdx.x * blockDim.x + threadIdx.x;
  if (idx >= (HC / 16) * KSTEPS * 32 * 16) return;   // 32768
  int e    = idx & 15;
  int lane = (idx >> 4) & 31;
  int ks   = (idx >> 9) & 3;
  int tn   = idx >> 11;
  int half = lane >> 4, l15 = lane & 15;
  int k = ks * 32 + ((e < 8) ? (half * 8 + e) : (16 + half * 8 + (e - 8)));
  int n = tn * 16 + l15;
  float f = W[k * HC + n];
  __bf16 hi = (__bf16)f;
  wsw_hi[idx] = hi;
  wsw_lo[idx] = (__bf16)(f - (float)hi);
}

// ---------- K2b: pre-swizzle x into A-operand bf16 hi/lo register layout ----------
// A layout (ISA 7.12.2): M = lane&15; e<8 -> K = kb + half*8 + e; e>=8 -> K = kb+16+half*8+(e-8)
__global__ void k_swizzle_x(const float* __restrict__ x,
                            __bf16* __restrict__ xsw_hi, __bf16* __restrict__ xsw_lo,
                            int M, int total) {
  int idx = blockIdx.x * blockDim.x + threadIdx.x;
  if (idx >= total) return;
  int e    = idx & 15;
  int lane = (idx >> 4) & 31;
  int ks   = (idx >> 9) & 3;
  int tm   = idx >> 11;
  int half = lane >> 4, l15 = lane & 15;
  int m = tm * 16 + l15;  if (m >= M) m = M - 1;
  int k = ks * 32 + ((e < 8) ? (half * 8 + e) : (16 + half * 8 + (e - 8)));
  float f = x[(size_t)m * NODE_DIM + k];
  __bf16 hi = (__bf16)f;
  xsw_hi[idx] = hi;
  xsw_lo[idx] = (__bf16)(f - (float)hi);
}

// ---------- K2c: h = x @ W via WMMA bf16x3, operands pre-swizzled ----------
// Inner loop: 8x global_load_b128 + 3x v_wmma per K-step.
__global__ __launch_bounds__(256) void k_gemm_h(const __bf16* __restrict__ xsw_hi,
                                                const __bf16* __restrict__ xsw_lo,
                                                const __bf16* __restrict__ wsw_hi,
                                                const __bf16* __restrict__ wsw_lo,
                                                float* __restrict__ h, int M) {
  int wave = (blockIdx.x * blockDim.x + threadIdx.x) >> 5;
  int lane = threadIdx.x & 31;
  const int tiles_n = HC / 16;                 // 16
  int tm = wave / tiles_n;
  int tn = wave % tiles_n;
  int mt = (M + 15) >> 4;
  if (tm >= mt) return;

  const v16bf* Ah = (const v16bf*)xsw_hi + (size_t)tm * KSTEPS * 32 + lane;
  const v16bf* Al = (const v16bf*)xsw_lo + (size_t)tm * KSTEPS * 32 + lane;
  const v16bf* Bh = (const v16bf*)wsw_hi + (size_t)tn * KSTEPS * 32 + lane;
  const v16bf* Bl = (const v16bf*)wsw_lo + (size_t)tn * KSTEPS * 32 + lane;

  v8f acc = {};
  #pragma unroll
  for (int ks = 0; ks < KSTEPS; ++ks) {
    v16bf ah = Ah[ks * 32];
    v16bf al = Al[ks * 32];
    v16bf bh = Bh[ks * 32];
    v16bf bl = Bl[ks * 32];
    acc = __builtin_amdgcn_wmma_f32_16x16x32_bf16(false, ah, false, bh, (short)0, acc, false, false);
    acc = __builtin_amdgcn_wmma_f32_16x16x32_bf16(false, ah, false, bl, (short)0, acc, false, false);
    acc = __builtin_amdgcn_wmma_f32_16x16x32_bf16(false, al, false, bh, (short)0, acc, false, false);
  }
  // C/D layout: VGPR v -> M = v + (lane>>4)*8, N = lane&15
  int half = lane >> 4, l15 = lane & 15;
  #pragma unroll
  for (int v = 0; v < 8; ++v) {
    int orow = tm * 16 + half * 8 + v;
    if (orow < M) h[(size_t)orow * HC + tn * 16 + l15] = acc[v];
  }
}

// ---------- K3: a_src/a_dst per (node, head): 64-wide dots on h ----------
__global__ __launch_bounds__(256) void k_node_attn(const float* __restrict__ h,
    const float* __restrict__ att_src, const float* __restrict__ att_dst,
    float* __restrict__ a_src, float* __restrict__ a_dst, int N) {
  int wave = (blockIdx.x * blockDim.x + threadIdx.x) >> 5;
  int lane = threadIdx.x & 31;
  if (wave >= N) return;
  int head = lane >> 3, sub = lane & 7;
  const float* hp = h + (size_t)wave * HC + lane * 8;    // = head*64 + sub*8
  const float* as = att_src + head * ODIM + sub * 8;
  const float* ad = att_dst + head * ODIM + sub * 8;
  float s = 0.f, d = 0.f;
  #pragma unroll
  for (int j = 0; j < 8; ++j) { float hv = hp[j]; s += hv * as[j]; d += hv * ad[j]; }
  #pragma unroll
  for (int m = 1; m < 8; m <<= 1) { s += __shfl_xor(s, m, 32); d += __shfl_xor(d, m, 32); }
  if (sub == 0) { a_src[wave * HEADS + head] = s; a_dst[wave * HEADS + head] = d; }
}

// ---------- K4: alpha = leakyrelu(a_src[s]+a_dst[d]+ea.V); segment max via atomicMax ----------
__global__ void k_edge_alpha(const int* __restrict__ src, const int* __restrict__ dst,
    const float* __restrict__ edge_attr, const float* __restrict__ loop_sum,
    const float* __restrict__ cnt, const float* __restrict__ v,
    const float* __restrict__ a_src, const float* __restrict__ a_dst,
    float* __restrict__ alpha, unsigned* __restrict__ amax_u, int E, int N) {
  int e = blockIdx.x * blockDim.x + threadIdx.x;
  if (e >= E + N) return;
  int s, d;
  float ea[EDIM];
  if (e < E) {
    s = src[e]; d = dst[e];
    #pragma unroll
    for (int i = 0; i < EDIM; ++i) ea[i] = edge_attr[e * EDIM + i];
  } else {
    s = d = e - E;
    float c = fmaxf(cnt[s], 1.0f);
    #pragma unroll
    for (int i = 0; i < EDIM; ++i) ea[i] = loop_sum[s * EDIM + i] / c;
  }
  #pragma unroll
  for (int hd = 0; hd < HEADS; ++hd) {
    float ae = 0.f;
    #pragma unroll
    for (int i = 0; i < EDIM; ++i) ae += ea[i] * v[i * HEADS + hd];
    float al = a_src[s * HEADS + hd] + a_dst[d * HEADS + hd] + ae;
    al = (al > 0.f) ? al : 0.2f * al;
    alpha[(size_t)e * HEADS + hd] = al;
    atomicMax(&amax_u[d * HEADS + hd], encf(al));
  }
}

// ---------- K5: expa = exp(alpha - amax[dst]); denom accumulation ----------
__global__ void k_edge_exp(const int* __restrict__ dst, const unsigned* __restrict__ amax_u,
                           float* __restrict__ alpha, float* __restrict__ denom, int E, int N) {
  int e = blockIdx.x * blockDim.x + threadIdx.x;
  if (e >= E + N) return;
  int d = (e < E) ? dst[e] : (e - E);
  #pragma unroll
  for (int hd = 0; hd < HEADS; ++hd) {
    float ex = __expf(alpha[(size_t)e * HEADS + hd] - decf(amax_u[d * HEADS + hd]));
    alpha[(size_t)e * HEADS + hd] = ex;
    atomicAdd(&denom[d * HEADS + hd], ex);
  }
}

// ---------- K6: msg scatter: out_sum[dst] += h[src] * att  (wave per edge) ----------
__global__ __launch_bounds__(256) void k_edge_scatter(const int* __restrict__ src,
    const int* __restrict__ dst, const float* __restrict__ h,
    const float* __restrict__ expa, const float* __restrict__ denom,
    float* __restrict__ out_sum, int E, int N) {
  int wave = (blockIdx.x * blockDim.x + threadIdx.x) >> 5;
  int lane = threadIdx.x & 31;
  if (wave >= E + N) return;
  int s, d;
  if (wave < E) { s = src[wave]; d = dst[wave]; } else { s = d = wave - E; }
  __builtin_prefetch(h + (size_t)s * HC + lane * 8, 0, 0);   // global_prefetch_b8 on gather row
  #pragma unroll
  for (int hd = 0; hd < HEADS; ++hd) {
    float att = expa[(size_t)wave * HEADS + hd] / (denom[d * HEADS + hd] + 1e-16f);
    const float2* hp = (const float2*)(h + (size_t)s * HC + hd * ODIM);
    float2 hv = hp[lane];
    float* op = out_sum + (size_t)d * HC + hd * ODIM + lane * 2;
    atomicAdd(op,     hv.x * att);
    atomicAdd(op + 1, hv.y * att);
  }
}

// ---------- K7: head mean + bias + LayerNorm + ELU (wave per node) ----------
__global__ __launch_bounds__(256) void k_finalize(const float* __restrict__ out_sum,
    const float* __restrict__ bias, const float* __restrict__ gamma,
    const float* __restrict__ beta, float* __restrict__ out, int N) {
  int wave = (blockIdx.x * blockDim.x + threadIdx.x) >> 5;
  int lane = threadIdx.x & 31;
  if (wave >= N) return;
  int c0 = lane * 2;
  float o0 = 0.f, o1 = 0.f;
  #pragma unroll
  for (int hd = 0; hd < HEADS; ++hd) {
    o0 += out_sum[(size_t)wave * HC + hd * ODIM + c0];
    o1 += out_sum[(size_t)wave * HC + hd * ODIM + c0 + 1];
  }
  o0 = o0 * 0.25f + bias[c0];
  o1 = o1 * 0.25f + bias[c0 + 1];
  float sum = o0 + o1, sq = o0 * o0 + o1 * o1;
  #pragma unroll
  for (int m = 1; m < 32; m <<= 1) { sum += __shfl_xor(sum, m, 32); sq += __shfl_xor(sq, m, 32); }
  float mu  = sum * (1.f / 64.f);
  float var = sq * (1.f / 64.f) - mu * mu;
  float inv = rsqrtf(var + 1e-5f);
  float r0 = (o0 - mu) * inv * gamma[c0]     + beta[c0];
  float r1 = (o1 - mu) * inv * gamma[c0 + 1] + beta[c0 + 1];
  r0 = (r0 > 0.f) ? r0 : (__expf(r0) - 1.f);
  r1 = (r1 > 0.f) ? r1 : (__expf(r1) - 1.f);
  out[(size_t)wave * ODIM + c0]     = r0;
  out[(size_t)wave * ODIM + c0 + 1] = r1;
}

extern "C" void kernel_launch(void* const* d_in, const int* in_sizes, int n_in,
                              void* d_out, int out_size, void* d_ws, size_t ws_size,
                              hipStream_t stream) {
  const float* x         = (const float*)d_in[0];
  const int*   edge_idx  = (const int*)  d_in[1];
  const float* edge_attr = (const float*)d_in[3];
  const float* W         = (const float*)d_in[4];
  const float* att_src   = (const float*)d_in[5];
  const float* att_dst   = (const float*)d_in[6];
  const float* W_e       = (const float*)d_in[7];
  const float* att_edge  = (const float*)d_in[8];
  const float* bias      = (const float*)d_in[9];
  const float* gamma     = (const float*)d_in[10];
  const float* beta      = (const float*)d_in[11];
  float*       out       = (float*)d_out;

  const int N  = in_sizes[0] / NODE_DIM;
  const int E  = in_sizes[3] / EDIM;
  const int Ep = E + N;
  const int mt = (N + 15) / 16;
  const int* src = edge_idx;
  const int* dst = edge_idx + E;

  // workspace carve-out (256B aligned)
  size_t off = 0;
  auto alloc = [&](size_t bytes) -> char* {
    char* p = (char*)d_ws + off;
    off += (bytes + 255) & ~(size_t)255;
    return p;
  };
  float*    h_buf    = (float*)   alloc((size_t)N * HC * 4);
  float*    out_sum  = (float*)   alloc((size_t)N * HC * 4);
  float*    loop_sum = (float*)   alloc((size_t)N * EDIM * 4);
  float*    cnt      = (float*)   alloc((size_t)N * 4);
  float*    a_src    = (float*)   alloc((size_t)N * HEADS * 4);
  float*    a_dst    = (float*)   alloc((size_t)N * HEADS * 4);
  float*    denom    = (float*)   alloc((size_t)N * HEADS * 4);
  unsigned* amax_u   = (unsigned*)alloc((size_t)N * HEADS * 4);
  float*    vbuf     = (float*)   alloc((size_t)EDIM * HEADS * 4);
  __bf16*   wsw_hi   = (__bf16*)  alloc((size_t)(HC / 16) * KSTEPS * 32 * 16 * 2);
  __bf16*   wsw_lo   = (__bf16*)  alloc((size_t)(HC / 16) * KSTEPS * 32 * 16 * 2);
  // xsw (hi+lo) is dead after the GEMM; alias alpha onto the same region.
  size_t xsw_elems   = (size_t)mt * KSTEPS * 32 * 16;        // per buffer (bf16)
  size_t shared_sz   = xsw_elems * 2 * 2;                    // hi+lo, 2B each
  size_t alpha_sz    = (size_t)Ep * HEADS * 4;
  char*  shared      = alloc(shared_sz > alpha_sz ? shared_sz : alpha_sz);
  __bf16* xsw_hi     = (__bf16*)shared;
  __bf16* xsw_lo     = xsw_hi + xsw_elems;
  float*  alpha      = (float*)shared;                       // valid only after K2c
  (void)ws_size; (void)n_in; (void)out_size;

  // zero accumulators (capture-safe async memsets)
  hipMemsetAsync(loop_sum, 0, (size_t)N * EDIM * 4, stream);
  hipMemsetAsync(cnt,      0, (size_t)N * 4, stream);
  hipMemsetAsync(denom,    0, (size_t)N * HEADS * 4, stream);
  hipMemsetAsync(amax_u,   0, (size_t)N * HEADS * 4, stream);   // enc-space minimum
  hipMemsetAsync(out_sum,  0, (size_t)N * HC * 4, stream);

  // K0: self-loop attr accumulation
  k_loop_accum<<<(E * EDIM + 255) / 256, 256, 0, stream>>>(edge_attr, dst, loop_sum, cnt, E);
  // K1: fold W_e @ att_edge -> V [16x4]
  k_precompute_v<<<1, EDIM * HEADS, 0, stream>>>(W_e, att_edge, vbuf);
  // K2a/K2b: operand pre-swizzle into WMMA register layouts (bf16 hi/lo)
  k_swizzle_w<<<((HC / 16) * KSTEPS * 32 * 16 + 255) / 256, 256, 0, stream>>>(W, wsw_hi, wsw_lo);
  k_swizzle_x<<<((int)xsw_elems + 255) / 256, 256, 0, stream>>>(x, xsw_hi, xsw_lo, N, (int)xsw_elems);
  // K2c: h = x @ W  (WMMA bf16x3; pure vmem+wmma inner loop)
  {
    int waves = mt * (HC / 16);
    k_gemm_h<<<(waves * 32 + 255) / 256, 256, 0, stream>>>(xsw_hi, xsw_lo, wsw_hi, wsw_lo, h_buf, N);
  }
  // K3: a_src / a_dst
  k_node_attn<<<((size_t)N * 32 + 255) / 256, 256, 0, stream>>>(h_buf, att_src, att_dst,
                                                                a_src, a_dst, N);
  // K4: alpha + segment max  (alpha aliases the now-dead xsw region)
  k_edge_alpha<<<(Ep + 255) / 256, 256, 0, stream>>>(src, dst, edge_attr, loop_sum, cnt,
                                                     vbuf, a_src, a_dst, alpha, amax_u, E, N);
  // K5: exp + denom
  k_edge_exp<<<(Ep + 255) / 256, 256, 0, stream>>>(dst, amax_u, alpha, denom, E, N);
  // K6: weighted scatter
  k_edge_scatter<<<((size_t)Ep * 32 + 255) / 256, 256, 0, stream>>>(src, dst, h_buf, alpha,
                                                                    denom, out_sum, E, N);
  // K7: head mean + LayerNorm + ELU
  k_finalize<<<((size_t)N * 32 + 255) / 256, 256, 0, stream>>>(out_sum, bias, gamma, beta, out, N);
}
